// AVT_VQVAE_Encoder_SingleTimestep_77034533421380
// MI455X (gfx1250) — compile-verified, matrix operand order checked by reading, not run on hardware
//
#include <hip/hip_runtime.h>
#include <hip/hip_bf16.h>
#include <cstdint>
#include <cstddef>

// ---------------------------------------------------------------------------
// AVT-VQVAE encoder forward for MI455X (gfx1250, wave32, WMMA + TDM).
//
// Dominant cost: 6x (4096x4096x2048) f16 GEMMs for the CMCM loss (~412 GFLOP)
// plus 3 encoder GEMMs and 3 pdist GEMMs. All matmuls run on
// v_wmma_f32_16x16x32_f16 with fp32 accumulation.
//   - scode GEMM stages its LDS tiles with the Tensor Data Mover
//     (tensor_load_to_lds + s_wait_tensorcnt), descriptor-padded to the
//     bank-conflict-free LDK=40 layout.
//   - pdist GEMM stages tiles with async global->LDS b128 copies
//     (global_load_async_to_lds_b128, ASYNCcnt-tracked).
// Intermediates (dist 32MB, scode 64MB) stay resident in the 192MB L2
// between the GEMM pass and the row-reduction pass.
// ---------------------------------------------------------------------------

typedef _Float16 h16;
typedef __attribute__((ext_vector_type(16))) _Float16 v16h;
typedef __attribute__((ext_vector_type(8)))  float    v8f;
typedef __attribute__((ext_vector_type(4)))  uint32_t u32x4;
typedef __attribute__((ext_vector_type(8)))  uint32_t u32x8;

#define AS3 __attribute__((address_space(3)))

#define BM 128
#define BN 64
#define BK 32           // == WMMA K for f16 -> one wmma per frag per k-step
#define LDK (BK + 8)    // padded LDS row stride (halfs) to dodge bank conflicts

// ---------------- CDNA5 async-tensor helpers --------------------------------

__device__ inline uint32_t lds_byte_offset(const void* p) {
  return (uint32_t)(uintptr_t)(const AS3 void*)p;
}

__device__ inline void wait_tensor0() {
#if __has_builtin(__builtin_amdgcn_s_wait_tensorcnt)
  __builtin_amdgcn_s_wait_tensorcnt(0);
#else
  asm volatile("s_wait_tensorcnt 0x0" ::: "memory");
#endif
}

__device__ inline void wait_async0() {
#if __has_builtin(__builtin_amdgcn_s_wait_asynccnt)
  __builtin_amdgcn_s_wait_asynccnt(0);
#else
  asm volatile("s_wait_asynccnt 0x0" ::: "memory");
#endif
}

// Async global->LDS 16B copy (VGLOBAL GV mode: VDST = LDS byte address,
// VADDR = 64-bit global address, SADDR = off). Tracked by ASYNCcnt.
__device__ inline void async_b128(const h16* g, h16* l) {
  uint32_t lo = lds_byte_offset(l);
  uint64_t ga = (uint64_t)(uintptr_t)g;
  asm volatile("global_load_async_to_lds_b128 %0, %1, off"
               :: "v"(lo), "v"(ga) : "memory");
}

// TDM: DMA a (tile1 x tile0) f16 tile out of a (td1 x td0) row-major f16
// tensor into LDS, inserting 4 DWORDs of padding after every 16 DWORDs
// (one 32-half row) -> reproduces the LDK=40 padded LDS layout.
__device__ inline void tdm_load_tile_f16(const h16* gsrc, uint32_t lds_off,
                                         uint32_t td0, uint32_t td1,
                                         uint32_t tile0, uint32_t tile1,
                                         uint32_t stride0_elems) {
  uint64_t ga = (uint64_t)(uintptr_t)gsrc;
  u32x4 g0;
  g0[0] = 1u;                                                  // count=1 (valid)
  g0[1] = lds_off;                                             // lds_addr (bytes)
  g0[2] = (uint32_t)ga;                                        // global_addr lo
  g0[3] = (uint32_t)((ga >> 32) & 0x01FFFFFFu) | (2u << 30);   // addr hi | type=2
  u32x8 g1;
  g1[0] = (1u << 16)        // data_size = 2 bytes
        | (1u << 20)        // pad_enable
        | (3u << 22)        // pad_interval code 3 -> every 16 DWORDs (64B row)
        | (3u << 25);       // pad_amount   code 3 -> 4 DWORDs (8 halfs)
  g1[1] = (td0 & 0xFFFFu) << 16;                   // tensor_dim0 lo16
  g1[2] = (td0 >> 16) | ((td1 & 0xFFFFu) << 16);   // tensor_dim0 hi | dim1 lo
  g1[3] = (td1 >> 16) | (tile0 << 16);             // tensor_dim1 hi | tile_dim0
  g1[4] = tile1 & 0xFFFFu;                         // tile_dim1 (tile_dim2 = 0)
  g1[5] = stride0_elems;                           // tensor_dim0_stride lo32
  g1[6] = 0u;                                      // stride hi | dim1_stride lo
  g1[7] = 0u;
  asm volatile("tensor_load_to_lds %0, %1" :: "s"(g0), "s"(g1) : "memory");
}

// ---------------- small utility kernels ------------------------------------

__global__ void k_zero_f32(float* p, int n) {
  int i = blockIdx.x * blockDim.x + threadIdx.x;
  if (i < n) p[i] = 0.0f;
}

__global__ void k_cast_f16(const float* __restrict__ x, h16* __restrict__ y, int n) {
  int i = blockIdx.x * blockDim.x + threadIdx.x;
  if (i < n) y[i] = (h16)x[i];
}

// out[r, 0:cols] = (f16) emb[r, coloff : coloff+cols] with row stride ld
__global__ void k_slice_cast(const float* __restrict__ emb, h16* __restrict__ out,
                             int cols, int ld, int coloff) {
  int r = blockIdx.x;
  for (int c = threadIdx.x; c < cols; c += blockDim.x)
    out[(size_t)r * cols + c] = (h16)emb[(size_t)r * ld + coloff + c];
}

// out[r] = sum_c x[r, coloff+c]^2  (fp32, one block per row)
__global__ void k_rownorm(const float* __restrict__ x, float* __restrict__ out,
                          int cols, int ld, int coloff) {
  __shared__ float red[256];
  int r = blockIdx.x;
  float s = 0.f;
  for (int c = threadIdx.x; c < cols; c += blockDim.x) {
    float v = x[(size_t)r * ld + coloff + c];
    s += v * v;
  }
  red[threadIdx.x] = s;
  __syncthreads();
  for (int w = 128; w > 0; w >>= 1) {
    if ((int)threadIdx.x < w) red[threadIdx.x] += red[threadIdx.x + w];
    __syncthreads();
  }
  if (threadIdx.x == 0) out[r] = red[0];
}

// ---------------- WMMA fragment loader --------------------------------------
// 16-bit A/B 16x32 fragment layout (ISA 7.12.2):
//   lane L: row = L&15, kbase = (L>>4)*8
//   v16h element pair (2i, 2i+1) holds K = (i>=4 ? 16:0) + kbase + 2*(i&3) (+1)
__device__ inline v16h frag_ld(const h16* __restrict__ tile, int rowbase, int lane) {
  const h16* p = tile + (size_t)(rowbase + (lane & 15)) * LDK;
  int kb = (lane >> 4) << 3;
  v16h f;
#pragma unroll
  for (int i = 0; i < 8; ++i) {
    int k = ((i & 4) << 2) + kb + ((i & 3) << 1);
    f[2 * i]     = p[k];
    f[2 * i + 1] = p[k + 1];
  }
  return f;
}

// C/D 16x16 f32 layout: lane L -> n = L&15, VGPR r -> m = r + 8*(L>>4)

// ---------------- encoder GEMM: H = relu(X @ W + b) -------------------------
// A: (4096, K) f16 row-major.  W: (K, N) f16 row-major (transposed into LDS).
__global__ __launch_bounds__(256)
void k_encoder_gemm(const h16* __restrict__ A, const h16* __restrict__ W,
                    const float* __restrict__ bias,
                    float* __restrict__ H, h16* __restrict__ Hh,
                    int K, int N) {
  __shared__ __align__(16) h16 As[BM][LDK];
  __shared__ __align__(16) h16 Bs[BN][LDK];
  const int tid = threadIdx.x;
  const int lane = tid & 31, wv = tid >> 5;
  const int wm = wv & 3, wn = wv >> 2;
  const int m0 = blockIdx.y * BM, n0 = blockIdx.x * BN;
  v8f acc[2][2] = {};

  for (int k0 = 0; k0 < K; k0 += BK) {
    { // A tile 128x32: each thread copies 16 contiguous halfs (2x b128)
      int row = tid >> 1, sg = tid & 1;
      const h16* src = A + (size_t)(m0 + row) * K + k0 + sg * 16;
      const uint4* s4 = (const uint4*)src;
      *(uint4*)&As[row][sg * 16]     = s4[0];
      *(uint4*)&As[row][sg * 16 + 8] = s4[1];
      if (k0 + BK < K) __builtin_prefetch(src + BK, 0, 1);
    }
    { // B tile: Bs[n][k] = W[k0+k][n0+n] (transpose from KxN)
      int n = tid & 63, ks = (tid >> 6) << 3;
#pragma unroll
      for (int kk = 0; kk < 8; ++kk)
        Bs[n][ks + kk] = W[(size_t)(k0 + ks + kk) * N + n0 + n];
    }
    __syncthreads();
    v16h af0 = frag_ld(&As[0][0], wm * 32,      lane);
    v16h af1 = frag_ld(&As[0][0], wm * 32 + 16, lane);
    v16h bf0 = frag_ld(&Bs[0][0], wn * 32,      lane);
    v16h bf1 = frag_ld(&Bs[0][0], wn * 32 + 16, lane);
    acc[0][0] = __builtin_amdgcn_wmma_f32_16x16x32_f16(false, af0, false, bf0, (short)0, acc[0][0], false, false);
    acc[0][1] = __builtin_amdgcn_wmma_f32_16x16x32_f16(false, af0, false, bf1, (short)0, acc[0][1], false, false);
    acc[1][0] = __builtin_amdgcn_wmma_f32_16x16x32_f16(false, af1, false, bf0, (short)0, acc[1][0], false, false);
    acc[1][1] = __builtin_amdgcn_wmma_f32_16x16x32_f16(false, af1, false, bf1, (short)0, acc[1][1], false, false);
    __syncthreads();
  }
  int nl = lane & 15, mo = (lane >> 4) << 3;
#pragma unroll
  for (int i = 0; i < 2; ++i)
#pragma unroll
    for (int j = 0; j < 2; ++j)
#pragma unroll
      for (int r = 0; r < 8; ++r) {
        int m = m0 + wm * 32 + i * 16 + mo + r;
        int n = n0 + wn * 32 + j * 16 + nl;
        float v = acc[i][j][r] + bias[n];
        v = v > 0.f ? v : 0.f;
        H [(size_t)m * N + n] = v;
        Hh[(size_t)m * N + n] = (h16)v;
      }
}

// ---------------- pdist GEMM: dist = ||x||^2 + ||e||^2 - 2 x e^T ------------
// A: (4096, K) f16.  E: (N, K) f16 row-major (NT GEMM).
// Tiles staged with async global->LDS b128 (ASYNCcnt).
__global__ __launch_bounds__(256)
void k_pdist_gemm(const h16* __restrict__ A, const h16* __restrict__ E,
                  const float* __restrict__ xn2, const float* __restrict__ en2,
                  float* __restrict__ dist, int K, int N) {
  __shared__ __align__(16) h16 As[BM][LDK];
  __shared__ __align__(16) h16 Bs[BN][LDK];
  const int tid = threadIdx.x;
  const int lane = tid & 31, wv = tid >> 5;
  const int wm = wv & 3, wn = wv >> 2;
  const int m0 = blockIdx.y * BM, n0 = blockIdx.x * BN;
  v8f acc[2][2] = {};

  for (int k0 = 0; k0 < K; k0 += BK) {
    { int row = tid >> 1, sg = tid & 1;
      const h16* src = A + (size_t)(m0 + row) * K + k0 + sg * 16;
      async_b128(src,     &As[row][sg * 16]);
      async_b128(src + 8, &As[row][sg * 16 + 8]);
      if (k0 + BK < K) __builtin_prefetch(src + BK, 0, 1);
    }
    { int n = tid >> 2, sg = tid & 3;
      async_b128(E + (size_t)(n0 + n) * K + k0 + sg * 8, &Bs[n][sg * 8]);
    }
    wait_async0();
    __syncthreads();
    v16h af0 = frag_ld(&As[0][0], wm * 32,      lane);
    v16h af1 = frag_ld(&As[0][0], wm * 32 + 16, lane);
    v16h bf0 = frag_ld(&Bs[0][0], wn * 32,      lane);
    v16h bf1 = frag_ld(&Bs[0][0], wn * 32 + 16, lane);
    acc[0][0] = __builtin_amdgcn_wmma_f32_16x16x32_f16(false, af0, false, bf0, (short)0, acc[0][0], false, false);
    acc[0][1] = __builtin_amdgcn_wmma_f32_16x16x32_f16(false, af0, false, bf1, (short)0, acc[0][1], false, false);
    acc[1][0] = __builtin_amdgcn_wmma_f32_16x16x32_f16(false, af1, false, bf0, (short)0, acc[1][0], false, false);
    acc[1][1] = __builtin_amdgcn_wmma_f32_16x16x32_f16(false, af1, false, bf1, (short)0, acc[1][1], false, false);
    __syncthreads();
  }
  int nl = lane & 15, mo = (lane >> 4) << 3;
#pragma unroll
  for (int i = 0; i < 2; ++i)
#pragma unroll
    for (int j = 0; j < 2; ++j)
#pragma unroll
      for (int r = 0; r < 8; ++r) {
        int m = m0 + wm * 32 + i * 16 + mo + r;
        int n = n0 + wn * 32 + j * 16 + nl;
        dist[(size_t)m * N + n] = xn2[m] + en2[n] - 2.0f * acc[i][j][r];
      }
}

// ---------------- scode GEMM: C = P @ Lq^T + Q @ Lp^T -----------------------
// Two K-segments of an NT GEMM (K=2048 each); LDS tiles staged by the Tensor
// Data Mover (one issue per workgroup from wave 0, TENSORcnt-tracked);
// fused per-block min for max_s.
__global__ __launch_bounds__(256)
void k_scode_gemm(const h16* __restrict__ A0, const h16* __restrict__ B0,
                  const h16* __restrict__ A1, const h16* __restrict__ B1,
                  float* __restrict__ C, float* __restrict__ partmin,
                  int K, int N) {
  __shared__ __align__(16) h16 As[BM][LDK];
  __shared__ __align__(16) h16 Bs[BN][LDK];
  __shared__ float red[256];
  const int tid = threadIdx.x;
  const int lane = tid & 31, wv = tid >> 5;
  const int wm = wv & 3, wn = wv >> 2;
  const int m0 = blockIdx.y * BM, n0 = blockIdx.x * BN;
  const uint32_t ldsA = lds_byte_offset(&As[0][0]);
  const uint32_t ldsB = lds_byte_offset(&Bs[0][0]);
  v8f acc[2][2] = {};

#pragma unroll 1
  for (int seg = 0; seg < 2; ++seg) {
    const h16* A = seg ? A1 : A0;
    const h16* B = seg ? B1 : B0;
#pragma unroll 1
    for (int k0 = 0; k0 < K; k0 += BK) {
      if (wv == 0) {
        // DMA the 128x32 A tile and 64x32 B tile into padded LDS.
        tdm_load_tile_f16(A + (size_t)m0 * K + k0, ldsA,
                          (uint32_t)K, 4096u, BK, BM, (uint32_t)K);
        tdm_load_tile_f16(B + (size_t)n0 * K + k0, ldsB,
                          (uint32_t)K, 4096u, BK, BN, (uint32_t)K);
        wait_tensor0();
      }
      __syncthreads();
      v16h af0 = frag_ld(&As[0][0], wm * 32,      lane);
      v16h af1 = frag_ld(&As[0][0], wm * 32 + 16, lane);
      v16h bf0 = frag_ld(&Bs[0][0], wn * 32,      lane);
      v16h bf1 = frag_ld(&Bs[0][0], wn * 32 + 16, lane);
      acc[0][0] = __builtin_amdgcn_wmma_f32_16x16x32_f16(false, af0, false, bf0, (short)0, acc[0][0], false, false);
      acc[0][1] = __builtin_amdgcn_wmma_f32_16x16x32_f16(false, af0, false, bf1, (short)0, acc[0][1], false, false);
      acc[1][0] = __builtin_amdgcn_wmma_f32_16x16x32_f16(false, af1, false, bf0, (short)0, acc[1][0], false, false);
      acc[1][1] = __builtin_amdgcn_wmma_f32_16x16x32_f16(false, af1, false, bf1, (short)0, acc[1][1], false, false);
      __syncthreads();
    }
  }
  int nl = lane & 15, mo = (lane >> 4) << 3;
  float tmin = 3.4e38f;
#pragma unroll
  for (int i = 0; i < 2; ++i)
#pragma unroll
    for (int j = 0; j < 2; ++j)
#pragma unroll
      for (int r = 0; r < 8; ++r) {
        int m = m0 + wm * 32 + i * 16 + mo + r;
        int n = n0 + wn * 32 + j * 16 + nl;
        float v = acc[i][j][r];
        C[(size_t)m * N + n] = v;
        tmin = fminf(tmin, v);
      }
  red[tid] = tmin;
  __syncthreads();
  for (int w = 128; w > 0; w >>= 1) {
    if (tid < w) red[tid] = fminf(red[tid], red[tid + w]);
    __syncthreads();
  }
  if (tid == 0) partmin[blockIdx.y * gridDim.x + blockIdx.x] = red[0];
}

// ---------------- per-row softmax(-sqrt(dist)) + argmin ---------------------
__global__ __launch_bounds__(256)
void k_row_softmax(const float* __restrict__ dist, h16* __restrict__ ph,
                   h16* __restrict__ logph, int* __restrict__ idx, int N) {
  __shared__ float sv[256];
  __shared__ int   si[256];
  const int i = blockIdx.x, tid = threadIdx.x;
  const float* row = dist + (size_t)i * N;
  float dloc[8];
  float bmin = 3.4e38f; int bidx = 0;
  const int nper = N / 256;   // 8
  for (int jj = 0; jj < nper; ++jj) {
    int j = tid + jj * 256;
    float d = row[j];
    dloc[jj] = d;
    if (d < bmin) { bmin = d; bidx = j; }
  }
  sv[tid] = bmin; si[tid] = bidx;
  __syncthreads();
  for (int w = 128; w > 0; w >>= 1) {
    if (tid < w) {
      float v2 = sv[tid + w]; int i2 = si[tid + w];
      if (v2 < sv[tid] || (v2 == sv[tid] && i2 < si[tid])) { sv[tid] = v2; si[tid] = i2; }
    }
    __syncthreads();
  }
  const float dmin = sv[0];
  const int   amin = si[0];
  const float smax = -sqrtf(fmaxf(dmin, 1e-12f));
  __syncthreads();

  float ev[8]; float ssum = 0.f;
  for (int jj = 0; jj < nper; ++jj) {
    float s = -sqrtf(fmaxf(dloc[jj], 1e-12f));
    float e = expf(s - smax);
    ev[jj] = e; ssum += e;
  }
  sv[tid] = ssum; __syncthreads();
  for (int w = 128; w > 0; w >>= 1) {
    if (tid < w) sv[tid] += sv[tid + w];
    __syncthreads();
  }
  const float inv = 1.0f / sv[0];
  for (int jj = 0; jj < nper; ++jj) {
    int j = tid + jj * 256;
    float p = ev[jj] * inv;
    ph   [(size_t)i * N + j] = (h16)p;
    logph[(size_t)i * N + j] = (h16)logf(p + 1e-10f);
  }
  if (tid == 0) idx[i] = amin;
}

// ---------------- gather outputs + commitment loss + bincount ---------------
__global__ __launch_bounds__(256)
void k_gather(const float* __restrict__ emb, const int* __restrict__ idx,
              const float* __restrict__ H, float* __restrict__ full_out,
              float* __restrict__ q_out, int coloff,
              float* __restrict__ loss_acc, int* __restrict__ counts) {
  __shared__ float red[256];
  const int i = blockIdx.x, tid = threadIdx.x;
  const int j = idx[i];
  const float* er = emb + (size_t)j * 3072;
  for (int c = tid; c < 3072; c += 256)
    full_out[(size_t)i * 3072 + c] = er[c];
  float s = 0.f;
  for (int c = tid; c < 1024; c += 256) {
    float e = er[coloff + c];
    q_out[(size_t)i * 1024 + c] = e;       // x + sg(e[idx]-x) == e[idx] numerically
    float d = H[(size_t)i * 1024 + c] - e;
    s += d * d;
  }
  red[tid] = s; __syncthreads();
  for (int w = 128; w > 0; w >>= 1) {
    if (tid < w) red[tid] += red[tid + w];
    __syncthreads();
  }
  if (tid == 0) { atomicAdd(loss_acc, red[0]); atomicAdd(&counts[j], 1); }
}

__global__ void k_perp(const int* __restrict__ counts, float* __restrict__ out_slot,
                       int M, float invB) {
  __shared__ float red[256];
  const int tid = threadIdx.x;
  float s = 0.f;
  for (int j = tid; j < M; j += 256) {
    float avg = (float)counts[j] * invB;
    s += avg * logf(avg + 1e-10f);
  }
  red[tid] = s; __syncthreads();
  for (int w = 128; w > 0; w >>= 1) {
    if (tid < w) red[tid] += red[tid + w];
    __syncthreads();
  }
  if (tid == 0) *out_slot = expf(-red[0]);
}

__global__ void k_equal(const int* __restrict__ a, const int* __restrict__ v,
                        const int* __restrict__ t, int* acc, int n) {
  int i = blockIdx.x * blockDim.x + threadIdx.x;
  if (i < n && a[i] == v[i] && a[i] == t[i]) atomicAdd(acc, 1);
}

__global__ void k_min_reduce(const float* __restrict__ pm, int n, float* maxs) {
  __shared__ float red[256];
  const int tid = threadIdx.x;
  float m = 3.4e38f;
  for (int i = tid; i < n; i += 256) m = fminf(m, pm[i]);
  red[tid] = m; __syncthreads();
  for (int w = 128; w > 0; w >>= 1) {
    if (tid < w) red[tid] = fminf(red[tid], red[tid + w]);
    __syncthreads();
  }
  if (tid == 0) *maxs = -red[0];   // max_s = max(-scode) = -min(scode)
}

// per-row: loss_i = -(scode_ii + max_s - log(sum_j exp(scode_ij + max_s) + EPS))
__global__ __launch_bounds__(256)
void k_rowpass(const float* __restrict__ C, const float* __restrict__ maxs_p,
               float* __restrict__ pair_loss, int N) {
  __shared__ float red[256];
  __shared__ float sdiag;
  const int i = blockIdx.x, tid = threadIdx.x;
  const float maxs = *maxs_p;
  const float* row = C + (size_t)i * N;
  float s = 0.f;
  for (int j = tid; j < N; j += 256) {
    float c = row[j];
    s += expf(c + maxs);
    if (j == i) sdiag = c;
  }
  red[tid] = s; __syncthreads();
  for (int w = 128; w > 0; w >>= 1) {
    if (tid < w) red[tid] += red[tid + w];
    __syncthreads();
  }
  if (tid == 0) {
    float pl = -(sdiag + maxs - logf(red[0] + 1e-5f));
    atomicAdd(pair_loss, pl * (1.0f / 4096.0f));
  }
}

__global__ void k_finalize(const float* __restrict__ scal, const int* __restrict__ equal_acc,
                           float* __restrict__ outsc) {
  if (blockIdx.x == 0 && threadIdx.x == 0) {
    const float sc = 0.25f / (4096.0f * 1024.0f);   // CC * mean
    outsc[0] = scal[0] * sc;                         // v_loss
    outsc[1] = scal[1] * sc;                         // a_loss
    outsc[2] = scal[2] * sc;                         // t_loss
    outsc[6] = (float)(*equal_acc);                  // equal_num
    outsc[7] = 0.5f * (scal[3] + scal[4] + scal[5]); // cmcm_loss
  }
}

// ---------------------------------------------------------------------------
extern "C" void kernel_launch(void* const* d_in, const int* in_sizes, int n_in,
                              void* d_out, int out_size, void* d_ws, size_t ws_size,
                              hipStream_t stream) {
  (void)in_sizes; (void)n_in; (void)out_size; (void)ws_size;

  const float* audio = (const float*)d_in[0];
  const float* video = (const float*)d_in[1];
  const float* text  = (const float*)d_in[2];
  const float* Wa = (const float*)d_in[3];
  const float* ba = (const float*)d_in[4];
  const float* Wv = (const float*)d_in[5];
  const float* bv = (const float*)d_in[6];
  const float* Wt = (const float*)d_in[7];
  const float* bt = (const float*)d_in[8];
  const float* emb = (const float*)d_in[9];
  float* out = (float*)d_out;

  // modality order m: 0=v, 1=a, 2=t (matches output order)
  const float* X32[3] = { video, audio, text };
  const float* W32[3] = { Wv, Wa, Wt };
  const float* B32[3] = { bv, ba, bt };
  const int    Kin[3]  = { 1024, 768, 768 };
  const int    coff[3] = { 0, 1024, 2048 };

  const size_t Bn = 4096, Dm = 1024, Mm = 2048;
  const size_t F = Bn * 3072, Q = Bn * Dm;      // full / q output sizes
  const size_t SCAL = 3 * (F + Q);              // scalar base in d_out

  // ---- workspace carve-up ----
  char* wsp = (char*)d_ws;
  size_t wo = 0;
  auto alloc_b = [&](size_t bytes) -> void* {
    void* p = wsp + wo;
    wo += (bytes + 255) & ~(size_t)255;
    return p;
  };
  h16*   Xh[3]; for (int m = 0; m < 3; ++m) Xh[m] = (h16*)alloc_b(Bn * Kin[m] * 2);
  h16*   Wh[3]; for (int m = 0; m < 3; ++m) Wh[m] = (h16*)alloc_b((size_t)Kin[m] * Dm * 2);
  h16*   Eh[3]; for (int m = 0; m < 3; ++m) Eh[m] = (h16*)alloc_b(Mm * Dm * 2);
  float* H[3];  for (int m = 0; m < 3; ++m) H[m]  = (float*)alloc_b(Bn * Dm * 4);
  h16*   Hh[3]; for (int m = 0; m < 3; ++m) Hh[m] = (h16*)alloc_b(Bn * Dm * 2);
  h16*   PH[3]; for (int m = 0; m < 3; ++m) PH[m] = (h16*)alloc_b(Bn * Mm * 2);
  h16*   LPH[3];for (int m = 0; m < 3; ++m) LPH[m]= (h16*)alloc_b(Bn * Mm * 2);
  float* dist   = (float*)alloc_b(Bn * Mm * 4);        // 32 MB (L2-resident)
  float* scode  = (float*)alloc_b(Bn * Bn * 4);        // 64 MB (L2-resident)
  float* xn2[3]; for (int m = 0; m < 3; ++m) xn2[m] = (float*)alloc_b(Bn * 4);
  float* en2[3]; for (int m = 0; m < 3; ++m) en2[m] = (float*)alloc_b(Mm * 4);
  int*   idx[3]; for (int m = 0; m < 3; ++m) idx[m] = (int*)alloc_b(Bn * 4);
  int*   cnt[3]; for (int m = 0; m < 3; ++m) cnt[m] = (int*)alloc_b(Mm * 4);
  float* partmin = (float*)alloc_b(2048 * 4);
  float* scal    = (float*)alloc_b(64 * 4);
  // scal slots: [0..2] loss_sq(v,a,t), [3..5] pair loss, [6] equal(int), [7] max_s

  // ---- zero accumulators ----
  k_zero_f32<<<1, 64, 0, stream>>>(scal, 64);
  for (int m = 0; m < 3; ++m)
    k_zero_f32<<<(int)((Mm + 255) / 256), 256, 0, stream>>>((float*)cnt[m], (int)Mm);

  // ---- casts to f16 ----
  for (int m = 0; m < 3; ++m) {
    int na = (int)(Bn * Kin[m]);
    k_cast_f16<<<(na + 255) / 256, 256, 0, stream>>>(X32[m], Xh[m], na);
    int nw = (int)((size_t)Kin[m] * Dm);
    k_cast_f16<<<(nw + 255) / 256, 256, 0, stream>>>(W32[m], Wh[m], nw);
    k_slice_cast<<<(int)Mm, 256, 0, stream>>>(emb, Eh[m], (int)Dm, 3072, coff[m]);
    k_rownorm<<<(int)Mm, 256, 0, stream>>>(emb, en2[m], (int)Dm, 3072, coff[m]);
  }

  // ---- encoders: H_m = relu(X_m @ W_m + b_m), also f16 copy ----
  for (int m = 0; m < 3; ++m) {
    dim3 g((unsigned)(Dm / BN), (unsigned)(Bn / BM));
    k_encoder_gemm<<<g, 256, 0, stream>>>(Xh[m], Wh[m], B32[m], H[m], Hh[m], Kin[m], (int)Dm);
    k_rownorm<<<(int)Bn, 256, 0, stream>>>(H[m], xn2[m], (int)Dm, (int)Dm, 0);
  }

  // ---- pdist + softmax/argmin per modality (dist buffer reused) ----
  for (int m = 0; m < 3; ++m) {
    dim3 g((unsigned)(Mm / BN), (unsigned)(Bn / BM));
    k_pdist_gemm<<<g, 256, 0, stream>>>(Hh[m], Eh[m], xn2[m], en2[m], dist, (int)Dm, (int)Mm);
    k_row_softmax<<<(int)Bn, 256, 0, stream>>>(dist, PH[m], LPH[m], idx[m], (int)Mm);
  }

  // ---- gathers (full + quantized outputs), commitment losses, bincount ----
  for (int m = 0; m < 3; ++m) {
    float* full_m = out + (size_t)m * (F + Q);
    float* q_m    = full_m + F;
    k_gather<<<(int)Bn, 256, 0, stream>>>(emb, idx[m], H[m], full_m, q_m,
                                          coff[m], scal + m, cnt[m]);
  }

  // ---- perplexities (write d_out directly) + equal count ----
  for (int m = 0; m < 3; ++m)
    k_perp<<<1, 256, 0, stream>>>(cnt[m], out + SCAL + 3 + m, (int)Mm, 1.0f / 4096.0f);
  k_equal<<<(int)(Bn / 256), 256, 0, stream>>>(idx[1], idx[0], idx[2],
                                               (int*)(scal + 6), (int)Bn);

  // ---- CMCM pairs: lcmcm(a,v), lcmcm(a,t), lcmcm(t,v) ----
  const int pairs[3][2] = { {1, 0}, {1, 2}, {2, 0} };
  for (int pi = 0; pi < 3; ++pi) {
    int p = pairs[pi][0], q = pairs[pi][1];
    dim3 g((unsigned)(Bn / BN), (unsigned)(Bn / BM));   // 64 x 32 blocks
    k_scode_gemm<<<g, 256, 0, stream>>>(PH[p], LPH[q], PH[q], LPH[p],
                                        scode, partmin, (int)Mm, (int)Bn);
    k_min_reduce<<<1, 256, 0, stream>>>(partmin, 2048, scal + 7);
    k_rowpass<<<(int)Bn, 256, 0, stream>>>(scode, scal + 7, scal + 3 + pi, (int)Bn);
  }

  // ---- scalars ----
  k_finalize<<<1, 64, 0, stream>>>(scal, (const int*)(scal + 6), out + SCAL);
}
